// NeuralRecursiveSystem_20916490731919
// MI455X (gfx1250) — matrix operand order repaired and checked by reading, not compiled, not run on hardware
//
#include <hip/hip_runtime.h>
#include <hip/hip_bf16.h>
#include <cstdint>

typedef _Float16 f16;
typedef __attribute__((ext_vector_type(16))) _Float16 v16h;
typedef __attribute__((ext_vector_type(8)))  _Float16 v8h;
typedef __attribute__((ext_vector_type(8)))  float    v8f;

#define N_    30000
#define E_    200000
#define TH_   1024
#define H_    8
#define C_    128
#define PROJ_ 256
#define OUT_  40
#define ESL_  (E_ + N_)
#define BN_EPS 1e-5f
#define TPW   5                 // M tiles (of 16) per wave: 30000 = 375 * (16*5)

// global-address-space pointer types: keep GEMM operand loads on the
// global path (global_load_b128) instead of flat_load (which also ties up
// DScnt / the LDS arbitration path per the CDNA5 ISA).
typedef __attribute__((address_space(1))) const f16  GCF16;
typedef __attribute__((address_space(1))) const v8h  GCV8H;
typedef __attribute__((address_space(1))) const v16h GCV16H;

// ---------------- device helpers ----------------
__device__ __forceinline__ float wave_sum(float v) {
  #pragma unroll
  for (int o = 16; o > 0; o >>= 1) v += __shfl_xor(v, o, 32);
  return v;
}

__device__ __forceinline__ void atomicMaxFloat(float* addr, float val) {
  if (val >= 0.f) atomicMax((int*)addr, __float_as_int(val));
  else            atomicMin((unsigned int*)addr, (unsigned int)__float_as_int(val));
}

// ---------------- WMMA GEMM ----------------
// B is pre-staged in fragment order: for tile (nt,kt), lane L, element j:
//   Bf[(((nt*KT)+kt)*32 + L)*16 + j] = B[kt*32 + 16*(L/16) + j][nt*16 + L%16]
// so each B fragment is ONE contiguous 32-byte load per lane.
// One wave computes an 80x16 strip (5 accumulators share each B fragment).
// grid = (NT, M/80), block = 32.
__global__ __launch_bounds__(32)
void gemm_wmma(const f16* __restrict__ A, int lda,
               const f16* __restrict__ Bf, int KT,
               const float* __restrict__ bias,
               float* __restrict__ C, int ldc, int Nreal) {
  const int lane = threadIdx.x;
  const int hs   = lane >> 4;              // half-select
  const int l16  = lane & 15;
  const int ncol = blockIdx.x * 16 + l16;  // C column for this lane
  const int m0   = blockIdx.y * (16 * TPW);
  const GCF16* ap[TPW];
  #pragma unroll
  for (int t = 0; t < TPW; ++t)
    ap[t] = (const GCF16*)(A + (size_t)(m0 + t * 16 + l16) * lda + hs * 16);
  const GCF16* bp = (const GCF16*)(Bf + (((size_t)blockIdx.x * KT) * 32 + lane) * 16);
  v8f acc[TPW] = {};
  for (int kt = 0; kt < KT; ++kt) {
    // issue all loads first so they clause together, then compute
    v16h b = *(const GCV16H*)bp;           // two global_load_b128
    bp += 32 * 16;
    v8h alo[TPW], ahi[TPW];
    #pragma unroll
    for (int t = 0; t < TPW; ++t) {
      alo[t] = *(const GCV8H*)(ap[t]);
      ahi[t] = *(const GCV8H*)(ap[t] + 8);
      ap[t] += 32;
    }
    #pragma unroll
    for (int t = 0; t < TPW; ++t) {
      v16h a;
      #pragma unroll
      for (int i = 0; i < 8; ++i) { a[i] = alo[t][i]; a[8 + i] = ahi[t][i]; }
      acc[t] = __builtin_amdgcn_wmma_f32_16x16x32_f16(false, a, false, b,
                                                      (short)0, acc[t], false, false);
    }
  }
  if (ncol < Nreal) {
    float bv = bias ? bias[ncol] : 0.f;
    #pragma unroll
    for (int t = 0; t < TPW; ++t) {
      #pragma unroll
      for (int r = 0; r < 8; ++r) {
        int cm = m0 + t * 16 + hs * 8 + r;
        C[(size_t)cm * ldc + ncol] = acc[t][r] + bv;
      }
    }
  }
}

// ---------------- staging / elementwise kernels ----------------
__global__ void fill_f32(float* p, float v, size_t n) {
  size_t i = (size_t)blockIdx.x * 256 + threadIdx.x;
  if (i < n) p[i] = v;
}

// row-major f32 -> row-major f16 with zero pad (used for GEMM A operands)
__global__ void cast_pad_f16(const float* __restrict__ s, f16* __restrict__ d,
                             int sr, int sc, int dr, int dc) {
  size_t i = (size_t)blockIdx.x * 256 + threadIdx.x;
  size_t tot = (size_t)dr * dc;
  if (i >= tot) return;
  int r = (int)(i / dc), c = (int)(i % dc);
  d[i] = (r < sr && c < sc) ? (f16)s[(size_t)r * sc + c] : (f16)0.f;
}

// row-major f32 weight -> WMMA B-fragment order f16 (zero pad K/N)
__global__ void stage_b_frag(const float* __restrict__ src, f16* __restrict__ dst,
                             int srcld, int K, int Nreal, int KT, int NT) {
  size_t i = (size_t)blockIdx.x * 256 + threadIdx.x;
  size_t tot = (size_t)KT * NT * 512;
  if (i >= tot) return;
  int j    = (int)(i & 15);
  int lane = (int)((i >> 4) & 31);
  size_t tile = i >> 9;
  int kt = (int)(tile % KT);
  int nt = (int)(tile / KT);
  int k = kt * 32 + ((lane >> 4) << 4) + j;
  int n = nt * 16 + (lane & 15);
  dst[i] = (k < K && n < Nreal) ? (f16)src[(size_t)k * srcld + n] : (f16)0.f;
}

// lp_w1 (2048x64) -> fragment-order f16 of the 1024x128 concat [w1a | w1b]
__global__ void stage_b_frag_lp(const float* __restrict__ src, f16* __restrict__ dst) {
  size_t i = (size_t)blockIdx.x * 256 + threadIdx.x;   // KT=32, NT=8
  if (i >= (size_t)32 * 8 * 512) return;
  int j    = (int)(i & 15);
  int lane = (int)((i >> 4) & 31);
  size_t tile = i >> 9;
  int kt = (int)(tile % 32);
  int nt = (int)(tile / 32);
  int k = kt * 32 + ((lane >> 4) << 4) + j;
  int n = nt * 16 + (lane & 15);
  float v = (n < 64) ? src[(size_t)k * 64 + n] : src[(size_t)(TH_ + k) * 64 + (n - 64)];
  dst[i] = (f16)v;
}

__global__ void stage_xfused(const float* __restrict__ xproj,
                             const float* __restrict__ emb,
                             const int* __restrict__ lab, f16* __restrict__ d) {
  size_t i = (size_t)blockIdx.x * 256 + threadIdx.x;
  if (i >= (size_t)N_ * TH_) return;
  int n = (int)(i >> 10), c = (int)(i & 1023);
  d[i] = (f16)(xproj[i] + emb[(size_t)lab[n] * TH_ + c]);
}

// ---------------- GAT attention ----------------
__global__ __launch_bounds__(128)
void att_dots(const float* __restrict__ xl, const float* __restrict__ aw,
              const float* __restrict__ dw, float* __restrict__ asrc,
              float* __restrict__ adst) {
  int wv = (int)(((size_t)blockIdx.x * 128 + threadIdx.x) >> 5);
  int lane = threadIdx.x & 31;
  if (wv >= N_ * H_) return;
  int n = wv >> 3, h = wv & 7;
  const float* xr = xl + (size_t)n * TH_ + h * C_;
  float s1 = 0.f, s2 = 0.f;
  for (int i = lane; i < C_; i += 32) {
    float v = xr[i];
    s1 += v * aw[h * C_ + i];
    s2 += v * dw[h * C_ + i];
  }
  s1 = wave_sum(s1); s2 = wave_sum(s2);
  if (lane == 0) { asrc[wv] = s1; adst[wv] = s2; }
}

__device__ __forceinline__ void edge_rc(const int* eidx, int e, int& row, int& col) {
  if (e < E_) { row = eidx[e]; col = eidx[E_ + e]; }
  else        { row = e - E_; col = e - E_; }
}

__global__ void edge_score_max(const int* __restrict__ eidx,
                               const float* __restrict__ asrc,
                               const float* __restrict__ adst,
                               float* __restrict__ alpha, float* __restrict__ segm) {
  size_t i = (size_t)blockIdx.x * 256 + threadIdx.x;
  if (i >= (size_t)ESL_ * H_) return;
  int e = (int)(i >> 3), h = (int)(i & 7);
  int row, col; edge_rc(eidx, e, row, col);
  float s = asrc[row * H_ + h] + adst[col * H_ + h];
  s = s > 0.f ? s : 0.2f * s;                 // leaky_relu 0.2
  alpha[i] = s;
  atomicMaxFloat(&segm[col * H_ + h], s);
}

__global__ void edge_exp_sum(const int* __restrict__ eidx,
                             float* __restrict__ alpha,
                             const float* __restrict__ segm, float* __restrict__ segz) {
  size_t i = (size_t)blockIdx.x * 256 + threadIdx.x;
  if (i >= (size_t)ESL_ * H_) return;
  int e = (int)(i >> 3), h = (int)(i & 7);
  int row, col; edge_rc(eidx, e, row, col); (void)row;
  float v = expf(alpha[i] - segm[col * H_ + h]);
  alpha[i] = v;
  atomicAdd(&segz[col * H_ + h], v);
}

// one block per edge, scatter xl[row]*alpha_norm into out[col]
__global__ __launch_bounds__(256)
void edge_aggregate(const int* __restrict__ eidx, const float* __restrict__ alpha,
                    const float* __restrict__ segz, const float* __restrict__ xl,
                    float* __restrict__ out) {
  int e = blockIdx.x;
  int row, col; edge_rc(eidx, e, row, col);
  __shared__ float w[H_];
  if (threadIdx.x < H_)
    w[threadIdx.x] = alpha[(size_t)e * H_ + threadIdx.x] /
                     (segz[col * H_ + threadIdx.x] + 1e-16f);
  __syncthreads();
  const float* xr = xl + (size_t)row * TH_;
  float* op = out + (size_t)col * TH_;
  for (int c = threadIdx.x; c < TH_; c += 256)
    atomicAdd(&op[c], xr[c] * w[c >> 7]);
}

// ---------------- BatchNorm ----------------
__global__ __launch_bounds__(256)
void bn_stats(const float* __restrict__ a, const float* __restrict__ b,
              float* __restrict__ sums, float* __restrict__ sumsq, int cols) {
  int c = blockIdx.x * 256 + threadIdx.x;
  int r0 = blockIdx.y * 512;
  int r1 = r0 + 512 < N_ ? r0 + 512 : N_;
  float s = 0.f, q = 0.f;
  for (int r = r0; r < r1; ++r) {
    float v = a[(size_t)r * cols + c];
    if (b) v += b[(size_t)r * cols + c];
    s += v; q += v * v;
  }
  atomicAdd(&sums[c], s); atomicAdd(&sumsq[c], q);
}

// out = act( g*(a[+b]-mu)*rsqrt(var+eps)+beta  [+res] [+emb[label]] )
__global__ void bn_apply(const float* __restrict__ a, const float* __restrict__ b,
                         const float* __restrict__ res, const float* __restrict__ embT,
                         const int* __restrict__ lab,
                         const float* __restrict__ g, const float* __restrict__ beta,
                         const float* __restrict__ sums, const float* __restrict__ sumsq,
                         float* __restrict__ out, int cols, int act) {
  size_t i = (size_t)blockIdx.x * 256 + threadIdx.x;
  if (i >= (size_t)N_ * cols) return;
  int c = (int)(i % cols), r = (int)(i / cols);
  float mu  = sums[c] * (1.f / N_);
  float var = sumsq[c] * (1.f / N_) - mu * mu;
  float v = a[i];
  if (b) v += b[i];
  v = g[c] * (v - mu) * rsqrtf(var + BN_EPS) + beta[c];
  if (res)  v += res[i];
  if (embT) v += embT[(size_t)lab[r] * cols + c];
  if (act == 1)      v = v > 0.f ? v : expf(v) - 1.f;   // elu
  else if (act == 2) v = v > 0.f ? v : 0.f;             // relu
  out[i] = v;
}

// ---------------- sampler ----------------
__global__ __launch_bounds__(128)
void edge_logits(const float* __restrict__ zw, const float* __restrict__ z,
                 const int* __restrict__ eidx, const float* __restrict__ biasp,
                 const float* __restrict__ gum, float* __restrict__ lraw,
                 float* __restrict__ samp) {
  int wv = (int)(((size_t)blockIdx.x * 128 + threadIdx.x) >> 5);
  int lane = threadIdx.x & 31;
  if (wv >= E_) return;
  int row = eidx[wv], col = eidx[E_ + wv];
  const float* ap = zw + (size_t)row * PROJ_;
  const float* bp = z  + (size_t)col * PROJ_;
  float s = 0.f;
  for (int i = lane; i < PROJ_; i += 32) s += ap[i] * bp[i];
  s = wave_sum(s);
  if (lane == 0) {
    float lr = fminf(10.f, fmaxf(-10.f, s + biasp[0]));
    lraw[wv] = lr;
    float g0 = gum[2 * wv], g1 = gum[2 * wv + 1];
    samp[wv] = (lr + g1 > g0) ? 1.f : 0.f;    // hard gumbel argmax (tie -> 0)
  }
}

__global__ void init_active(const unsigned char* __restrict__ m, float* __restrict__ act) {
  int i = blockIdx.x * 256 + threadIdx.x;
  if (i < N_) act[i] = m[i] ? 1.f : 0.f;
}

__global__ void hop_edge(const int* __restrict__ eidx, const float* __restrict__ samp,
                         const float* __restrict__ act, float* __restrict__ fw,
                         float* __restrict__ na) {
  int e = blockIdx.x * 256 + threadIdx.x;
  if (e >= E_) return;
  int row = eidx[e], col = eidx[E_ + e];
  float sw = samp[e] * act[col];
  if (sw > fw[e]) fw[e] = sw;
  if (sw != 0.f) atomicAdd(&na[row], sw);
}

__global__ void thresh_active(const float* __restrict__ na, float* __restrict__ act) {
  int i = blockIdx.x * 256 + threadIdx.x;
  if (i < N_) act[i] = na[i] > 1e-5f ? 1.f : 0.f;
}

__global__ void nw_max(const int* __restrict__ eidx, const float* __restrict__ fw,
                       float* __restrict__ m2) {
  int e = blockIdx.x * 256 + threadIdx.x;
  if (e < E_) atomicMaxFloat(&m2[eidx[E_ + e]], fw[e]);
}
__global__ void nw_exp(const int* __restrict__ eidx, const float* __restrict__ fw,
                       const float* __restrict__ m2, float* __restrict__ nw,
                       float* __restrict__ z2) {
  int e = blockIdx.x * 256 + threadIdx.x;
  if (e >= E_) return;
  float v = expf(fw[e] - m2[eidx[E_ + e]]);
  nw[e] = v; atomicAdd(&z2[eidx[E_ + e]], v);
}
__global__ void nw_div(const int* __restrict__ eidx, float* __restrict__ nw,
                       const float* __restrict__ z2) {
  int e = blockIdx.x * 256 + threadIdx.x;
  if (e < E_) nw[e] = nw[e] / (z2[eidx[E_ + e]] + 1e-16f);
}

__global__ __launch_bounds__(256)
void scatter_msg(const int* __restrict__ eidx, const float* __restrict__ nw,
                 const float* __restrict__ hl, float* __restrict__ msg) {
  int e = blockIdx.x;
  float w = nw[e];
  if (w == 0.f) return;
  int row = eidx[e], col = eidx[E_ + e];
  const float* hr = hl + (size_t)row * TH_;
  float* mp = msg + (size_t)col * TH_;
  for (int c = threadIdx.x; c < TH_; c += 256) atomicAdd(&mp[c], w * hr[c]);
}

// ---------------- heads ----------------
__global__ void logsoftmax40(const float* __restrict__ in, float* __restrict__ out) {
  int n = blockIdx.x * 256 + threadIdx.x;
  if (n >= N_) return;
  const float* r = in + (size_t)n * OUT_;
  float mx = -1e30f;
  for (int j = 0; j < OUT_; ++j) mx = fmaxf(mx, r[j]);
  float s = 0.f;
  for (int j = 0; j < OUT_; ++j) s += expf(r[j] - mx);
  float ls = logf(s);
  float* o = out + (size_t)n * OUT_;
  for (int j = 0; j < OUT_; ++j) o[j] = r[j] - mx - ls;
}

__global__ __launch_bounds__(128)
void link_pred(const float* __restrict__ p, const int* __restrict__ eidx,
               const float* __restrict__ b1, const float* __restrict__ w2,
               const float* __restrict__ b2, float* __restrict__ link) {
  int wv = (int)(((size_t)blockIdx.x * 128 + threadIdx.x) >> 5);
  int lane = threadIdx.x & 31;
  if (wv >= E_) return;
  int row = eidx[wv], col = eidx[E_ + wv];
  float acc = 0.f;
  #pragma unroll
  for (int t = 0; t < 2; ++t) {
    int j = lane * 2 + t;
    float v = p[(size_t)row * 128 + j] + p[(size_t)col * 128 + 64 + j] + b1[j];
    v = v > 0.f ? v : 0.f;
    acc += v * w2[j];
  }
  acc = wave_sum(acc);
  if (lane == 0) link[wv] = 1.f / (1.f + expf(-(acc + b2[0])));
}

// ---------------- host orchestration ----------------
static inline size_t cdiv(size_t a, size_t b) { return (a + b - 1) / b; }

extern "C" void kernel_launch(void* const* d_in, const int* in_sizes, int n_in,
                              void* d_out, int out_size, void* d_ws, size_t ws_size,
                              hipStream_t stream) {
  const float* x      = (const float*)d_in[0];
  const int*   eidx   = (const int*)d_in[1];
  const int*   ylab   = (const int*)d_in[2];
  const unsigned char* tmask = (const unsigned char*)d_in[3];
  const float* gum    = (const float*)d_in[4];
  const float* res_w  = (const float*)d_in[5];
  const float* res_b  = (const float*)d_in[6];
  const float* emb    = (const float*)d_in[7];
  const float* g1_w   = (const float*)d_in[8];
  const float* g1_as  = (const float*)d_in[9];
  const float* g1_ad  = (const float*)d_in[10];
  /* gat1_b d_in[11] cancels through BN */
  const float* bn1_g  = (const float*)d_in[12];
  const float* bn1_b  = (const float*)d_in[13];
  const float* g2_w   = (const float*)d_in[14];
  const float* g2_as  = (const float*)d_in[15];
  const float* g2_ad  = (const float*)d_in[16];
  /* gat2_b d_in[17] cancels through BN */
  const float* bn2_g  = (const float*)d_in[18];
  const float* bn2_b  = (const float*)d_in[19];
  const float* samp_pw = (const float*)d_in[20];
  const float* samp_pb = (const float*)d_in[21];
  const float* samp_W  = (const float*)d_in[22];
  const float* samp_bias = (const float*)d_in[23];
  const float* enh_w  = (const float*)d_in[24];
  const float* enh_b  = (const float*)d_in[25];
  const float* bne_g  = (const float*)d_in[26];
  const float* bne_b  = (const float*)d_in[27];
  const float* lp_w1  = (const float*)d_in[28];
  const float* lp_b1  = (const float*)d_in[29];
  const float* lp_w2  = (const float*)d_in[30];
  const float* lp_b2  = (const float*)d_in[31];
  const float* cls_w1 = (const float*)d_in[32];
  const float* cls_b1 = (const float*)d_in[33];
  const float* cbn_g  = (const float*)d_in[34];
  const float* cbn_b  = (const float*)d_in[35];
  const float* cls_w2 = (const float*)d_in[36];
  const float* cls_b2 = (const float*)d_in[37];

  // ---- workspace carve ----
  char* ws = (char*)d_ws; size_t o = 0;
  auto take = [&](size_t bytes) -> void* {
    void* p = ws + o; o += (bytes + 255) & ~(size_t)255; return p;
  };
  float* xproj = (float*)take((size_t)N_ * TH_ * 4);
  float* gemo  = (float*)take((size_t)N_ * TH_ * 4);   // xl / hl / cls hidden
  float* aggo  = (float*)take((size_t)N_ * TH_ * 4);   // GAT agg / msg / relu(c)
  float* hbuf  = (float*)take((size_t)N_ * TH_ * 4);   // h1 -> h_base
  float* zbuf  = (float*)take((size_t)N_ * PROJ_ * 4); // z  / cls logits
  float* zwbuf = (float*)take((size_t)N_ * PROJ_ * 4); // zw / lp "p"
  f16*   A16   = (f16*)take((size_t)N_ * TH_ * 2);
  f16*   W16   = (f16*)take((size_t)TH_ * TH_ * 2);    // fragment-order weights
  float* asrc  = (float*)take((size_t)N_ * H_ * 4);
  float* adst  = (float*)take((size_t)N_ * H_ * 4);
  float* alpha = (float*)take((size_t)ESL_ * H_ * 4);
  float* segm  = (float*)take((size_t)N_ * H_ * 4);
  float* segz  = (float*)take((size_t)N_ * H_ * 4);
  float* sums  = (float*)take(TH_ * 4);
  float* sumsq = (float*)take(TH_ * 4);
  float* samp  = (float*)take((size_t)E_ * 4);
  float* activ = (float*)take((size_t)N_ * 4);
  float* newact= (float*)take((size_t)N_ * 4);
  float* m2    = (float*)take((size_t)N_ * 4);
  float* z2    = (float*)take((size_t)N_ * 4);
  float* nw    = (float*)take((size_t)E_ * 4);

  // ---- output carve: (log_probs, link, h_enh, final_w, logits_raw) ----
  float* out    = (float*)d_out;
  float* o_logp = out;
  float* o_link = o_logp + (size_t)N_ * OUT_;
  float* o_henh = o_link + E_;
  float* o_fw   = o_henh + (size_t)N_ * TH_;
  float* o_lraw = o_fw + E_;

  const size_t NT = (size_t)N_ * TH_;
  auto stageB = [&](const float* src, int srcld, int K, int Nreal, int KT, int NTt) {
    size_t tot = (size_t)KT * NTt * 512;
    stage_b_frag<<<(unsigned)cdiv(tot, 256), 256, 0, stream>>>(src, W16, srcld, K, Nreal, KT, NTt);
  };
  auto gemm = [&](const f16* A, int lda, int KT, const float* bias,
                  float* C, int ldc, int Nreal) {
    dim3 g((unsigned)cdiv(Nreal, 16), N_ / (16 * TPW));   // 375 strips of 80 rows
    gemm_wmma<<<g, 32, 0, stream>>>(A, lda, W16, KT, bias, C, ldc, Nreal);
  };
  auto stats = [&](const float* a, const float* b, int cols) {
    (void)hipMemsetAsync(sums, 0, TH_ * 4, stream);
    (void)hipMemsetAsync(sumsq, 0, TH_ * 4, stream);
    dim3 g(cols / 256, (unsigned)cdiv(N_, 512));
    bn_stats<<<g, 256, 0, stream>>>(a, b, sums, sumsq, cols);
  };
  auto gat_layer = [&](const float* a_s, const float* a_d) {
    att_dots<<<(unsigned)cdiv((size_t)N_ * H_ * 32, 128), 128, 0, stream>>>(gemo, a_s, a_d, asrc, adst);
    fill_f32<<<(unsigned)cdiv((size_t)N_ * H_, 256), 256, 0, stream>>>(segm, -INFINITY, (size_t)N_ * H_);
    (void)hipMemsetAsync(segz, 0, (size_t)N_ * H_ * 4, stream);
    edge_score_max<<<(unsigned)cdiv((size_t)ESL_ * H_, 256), 256, 0, stream>>>(eidx, asrc, adst, alpha, segm);
    edge_exp_sum<<<(unsigned)cdiv((size_t)ESL_ * H_, 256), 256, 0, stream>>>(eidx, alpha, segm, segz);
    (void)hipMemsetAsync(aggo, 0, NT * 4, stream);
    edge_aggregate<<<ESL_, 256, 0, stream>>>(eidx, alpha, segz, gemo, aggo);
  };

  // ===== 1) x_proj = x @ res_w + res_b (K padded 500->512) =====
  cast_pad_f16<<<(unsigned)cdiv((size_t)N_ * 512, 256), 256, 0, stream>>>(x, A16, N_, 500, N_, 512);
  stageB(res_w, TH_, 500, TH_, 16, 64);
  gemm(A16, 512, 16, res_b, xproj, TH_, TH_);

  // ===== 2) GAT layer 1 on x_fused = x_proj + emb[y] =====
  stage_xfused<<<(unsigned)cdiv(NT, 256), 256, 0, stream>>>(xproj, emb, ylab, A16);
  stageB(g1_w, TH_, TH_, TH_, 32, 64);
  gemm(A16, TH_, 32, nullptr, gemo, TH_, TH_);
  gat_layer(g1_as, g1_ad);
  stats(aggo, nullptr, TH_);
  bn_apply<<<(unsigned)cdiv(NT, 256), 256, 0, stream>>>(aggo, nullptr, nullptr, emb, ylab,
      bn1_g, bn1_b, sums, sumsq, hbuf, TH_, 1);                 // h1 (elu, +y_emb)

  // ===== 3) GAT layer 2 =====
  cast_pad_f16<<<(unsigned)cdiv(NT, 256), 256, 0, stream>>>(hbuf, A16, N_, TH_, N_, TH_);
  stageB(g2_w, TH_, TH_, TH_, 32, 64);
  gemm(A16, TH_, 32, nullptr, gemo, TH_, TH_);
  gat_layer(g2_as, g2_ad);
  stats(aggo, nullptr, TH_);
  bn_apply<<<(unsigned)cdiv(NT, 256), 256, 0, stream>>>(aggo, nullptr, xproj, nullptr, nullptr,
      bn2_g, bn2_b, sums, sumsq, hbuf, TH_, 1);                 // h_base (elu, +x_proj)

  // ===== 4) sampler: z = h_base@samp_pw+pb ; zw = z@samp_W ; edge logits =====
  cast_pad_f16<<<(unsigned)cdiv(NT, 256), 256, 0, stream>>>(hbuf, A16, N_, TH_, N_, TH_);
  stageB(samp_pw, PROJ_, TH_, PROJ_, 32, 16);
  gemm(A16, TH_, 32, samp_pb, zbuf, PROJ_, PROJ_);
  cast_pad_f16<<<(unsigned)cdiv((size_t)N_ * PROJ_, 256), 256, 0, stream>>>(zbuf, A16, N_, PROJ_, N_, PROJ_);
  stageB(samp_W, PROJ_, PROJ_, PROJ_, 8, 16);
  gemm(A16, PROJ_, 8, nullptr, zwbuf, PROJ_, PROJ_);
  edge_logits<<<(unsigned)cdiv((size_t)E_ * 32, 128), 128, 0, stream>>>(zwbuf, zbuf, eidx, samp_bias, gum, o_lraw, samp);

  // ===== 5) 2-hop propagation -> final_w, nw =====
  init_active<<<(unsigned)cdiv(N_, 256), 256, 0, stream>>>(tmask, activ);
  (void)hipMemsetAsync(o_fw, 0, (size_t)E_ * 4, stream);
  for (int hop = 0; hop < 2; ++hop) {
    (void)hipMemsetAsync(newact, 0, (size_t)N_ * 4, stream);
    hop_edge<<<(unsigned)cdiv(E_, 256), 256, 0, stream>>>(eidx, samp, activ, o_fw, newact);
    if (hop == 0)
      thresh_active<<<(unsigned)cdiv(N_, 256), 256, 0, stream>>>(newact, activ);
  }
  fill_f32<<<(unsigned)cdiv(N_, 256), 256, 0, stream>>>(m2, -INFINITY, N_);
  (void)hipMemsetAsync(z2, 0, (size_t)N_ * 4, stream);
  nw_max<<<(unsigned)cdiv(E_, 256), 256, 0, stream>>>(eidx, o_fw, m2);
  nw_exp<<<(unsigned)cdiv(E_, 256), 256, 0, stream>>>(eidx, o_fw, m2, nw, z2);
  nw_div<<<(unsigned)cdiv(E_, 256), 256, 0, stream>>>(eidx, nw, z2);

  // ===== 6) enhancement x2: h = elu(bn(h + seg_sum(nw*(h@enh_w+enh_b)[row]))) =====
  stageB(enh_w, TH_, TH_, TH_, 32, 64);
  const float* h_cur = hbuf;
  for (int it = 0; it < 2; ++it) {
    cast_pad_f16<<<(unsigned)cdiv(NT, 256), 256, 0, stream>>>(h_cur, A16, N_, TH_, N_, TH_);
    gemm(A16, TH_, 32, enh_b, gemo, TH_, TH_);                  // hl
    (void)hipMemsetAsync(aggo, 0, NT * 4, stream);
    scatter_msg<<<E_, 256, 0, stream>>>(eidx, nw, gemo, aggo);  // msg
    stats(h_cur, aggo, TH_);
    bn_apply<<<(unsigned)cdiv(NT, 256), 256, 0, stream>>>(h_cur, aggo, nullptr, nullptr, nullptr,
        bne_g, bne_b, sums, sumsq, o_henh, TH_, 1);
    h_cur = o_henh;
  }

  // ===== 7) classifier =====
  cast_pad_f16<<<(unsigned)cdiv(NT, 256), 256, 0, stream>>>(o_henh, A16, N_, TH_, N_, TH_);
  stageB(cls_w1, 512, TH_, 512, 32, 32);
  gemm(A16, TH_, 32, cls_b1, gemo, 512, 512);
  stats(gemo, nullptr, 512);
  bn_apply<<<(unsigned)cdiv((size_t)N_ * 512, 256), 256, 0, stream>>>(gemo, nullptr, nullptr, nullptr, nullptr,
      cbn_g, cbn_b, sums, sumsq, aggo, 512, 2);                 // relu
  cast_pad_f16<<<(unsigned)cdiv((size_t)N_ * 512, 256), 256, 0, stream>>>(aggo, A16, N_, 512, N_, 512);
  stageB(cls_w2, OUT_, 512, OUT_, 16, 3);
  gemm(A16, 512, 16, cls_b2, zbuf, OUT_, OUT_);
  logsoftmax40<<<(unsigned)cdiv(N_, 256), 256, 0, stream>>>(zbuf, o_logp);

  // ===== 8) link predictor: p = h_enh @ [w1a|w1b] (1024x128) =====
  cast_pad_f16<<<(unsigned)cdiv(NT, 256), 256, 0, stream>>>(o_henh, A16, N_, TH_, N_, TH_);
  stage_b_frag_lp<<<(unsigned)cdiv((size_t)32 * 8 * 512, 256), 256, 0, stream>>>(lp_w1, W16);
  gemm(A16, TH_, 32, nullptr, zwbuf, 128, 128);
  link_pred<<<(unsigned)cdiv((size_t)E_ * 32, 128), 128, 0, stream>>>(zwbuf, eidx, lp_b1, lp_w2, lp_b2, o_link);
}
